// MaskAwareTransformer_10187662426280
// MI455X (gfx1250) — compile-verified, matrix operand order checked by reading, not run on hardware
//
#include <hip/hip_runtime.h>
#include <hip/hip_bf16.h>
#include <type_traits>

// ---------- CDNA5 WMMA types ----------
typedef __attribute__((ext_vector_type(16))) __bf16 v16bf;
typedef __attribute__((ext_vector_type(8)))  float  v8f;
typedef __attribute__((ext_vector_type(4)))  unsigned int u32x4;
typedef __attribute__((ext_vector_type(8)))  int i32x8;
typedef __attribute__((ext_vector_type(4)))  int i32x4;

union FragU { v16bf v; uint4 q[2]; };

#if defined(__has_builtin)
#if __has_builtin(__builtin_amdgcn_tensor_load_to_lds) && \
    __has_builtin(__builtin_amdgcn_s_wait_tensorcnt)
#define USE_TDM 1
#endif
#endif
#ifndef USE_TDM
#define USE_TDM 0
#endif

__device__ __forceinline__ v8f wmma_bf16(v16bf a, v16bf b, v8f c) {
  return __builtin_amdgcn_wmma_f32_16x16x32_bf16(false, a, false, b,
                                                 (short)0, c, false, false);
}

// 16x32 bf16 fragment from LDS tile: lane l -> row l&15, chunks 8*(l>>4), +16
__device__ __forceinline__ v16bf ld_frag(const __bf16* tile, int stride) {
  int lane = threadIdx.x & 31;
  int r  = lane & 15;
  int c0 = (lane >> 4) << 3;
  FragU u;
  const __bf16* p = tile + r * stride + c0;
  u.q[0] = *reinterpret_cast<const uint4*>(p);
  u.q[1] = *reinterpret_cast<const uint4*>(p + 16);
  return u.v;
}

// LDS byte offset: flat shared address keeps the LDS offset in the low 32 bits
__device__ __forceinline__ unsigned lds_off(const void* p) {
  return (unsigned)(unsigned long long)(uintptr_t)p;
}

// ---- Tensor Data Mover: 2D bf16 tile (tile_d1 rows x tile_d0 elems) ----
__device__ __forceinline__ void tdm_load_2d_bf16(
    unsigned lds_byte, const void* gptr,
    unsigned tensor_d0, unsigned tensor_d1, unsigned stride0,
    unsigned tile_d0, unsigned tile_d1) {
#if USE_TDM
  unsigned long long ga = (unsigned long long)(uintptr_t)gptr;
  u32x4 g0;
  i32x8 g1;
  i32x4 g2 = {0, 0, 0, 0};
  i32x4 g3 = {0, 0, 0, 0};
  i32x8 g4 = {0, 0, 0, 0, 0, 0, 0, 0};
  g0[0] = 1u;                                        // count=1, user D#
  g0[1] = lds_byte;                                  // lds_addr
  g0[2] = (unsigned)ga;                              // global_addr[31:0]
  g0[3] = (unsigned)((ga >> 32) & 0x1FFFFFFu) | (2u << 30);  // addr hi | type=2
  g1[0] = (int)(1u << 16);                           // data_size=1 (2 bytes)
  g1[1] = (int)((tensor_d0 & 0xFFFFu) << 16);        // tensor_dim0 lo
  g1[2] = (int)(((tensor_d0 >> 16) & 0xFFFFu) | ((tensor_d1 & 0xFFFFu) << 16));
  g1[3] = (int)(((tensor_d1 >> 16) & 0xFFFFu) | ((tile_d0 & 0xFFFFu) << 16));
  g1[4] = (int)(tile_d1 & 0xFFFFu);                  // tile_dim1 (tile_dim2=0)
  g1[5] = (int)stride0;                              // tensor_dim0_stride lo32
  g1[6] = 0;                                         // stride hi16 | dim1_stride
  g1[7] = 0;
  __builtin_amdgcn_tensor_load_to_lds(g0, g1, g2, g3, g4, 0);
#endif
}

__device__ __forceinline__ void tdm_wait() {
#if USE_TDM
  __builtin_amdgcn_s_wait_tensorcnt(0);
#endif
}

// ---------- problem constants ----------
#define BQ   4
#define CCH  3
#define HW   256
#define PP   8
#define NP   1024
#define DD   512
#define DFF  2048
#define NHH  8
#define DKK  64
#define LL   8
#define MM   (BQ * NP)

// =====================================================================
// f32 -> bf16 bulk convert (weights pre-pass); n multiple of 4
// =====================================================================
__global__ __launch_bounds__(256) void cvt_bf16_kernel(
    const float* __restrict__ src, __bf16* __restrict__ dst, int n4) {
  int i = blockIdx.x * 256 + threadIdx.x;
  if (i >= n4) return;
  float4 v = reinterpret_cast<const float4*>(src)[i];
  union { __bf16 h[4]; uint2 u; } p;
  p.h[0] = (__bf16)v.x; p.h[1] = (__bf16)v.y;
  p.h[2] = (__bf16)v.z; p.h[3] = (__bf16)v.w;
  reinterpret_cast<uint2*>(dst)[i] = p.u;
}

// =====================================================================
// Patch + mask embedding + positional encoding (one thread per h[m][d])
// =====================================================================
__global__ __launch_bounds__(256) void embed_kernel(
    const float* __restrict__ x, const float* __restrict__ mask,
    const float* __restrict__ pw, const float* __restrict__ pb,
    const float* __restrict__ mw, const float* __restrict__ mb,
    float* __restrict__ h) {
  size_t idx = (size_t)blockIdx.x * 256 + threadIdx.x;
  int d = (int)(idx & (DD - 1));
  int m = (int)(idx >> 9);
  int b = m >> 10, n = m & (NP - 1);
  int pr = n >> 5, pc = n & 31;

  const float* wr = pw + (size_t)d * (CCH * PP * PP);
  float acc = pb[d] + mb[d];
#pragma unroll
  for (int c = 0; c < CCH; ++c)
    for (int py = 0; py < PP; ++py)
#pragma unroll
      for (int px = 0; px < PP; ++px)
        acc += x[(((size_t)(b * CCH + c) * HW) + pr * PP + py) * HW + pc * PP + px]
             * wr[(c * PP + py) * PP + px];

  const float* mr = mw + (size_t)d * (PP * PP);
  for (int py = 0; py < PP; ++py)
#pragma unroll
    for (int px = 0; px < PP; ++px)
      acc += mask[(((size_t)b * HW) + pr * PP + py) * HW + pc * PP + px]
           * mr[py * PP + px];

  int i = d >> 1;
  float div = __expf((float)(2 * i) * (-9.210340371976184f / (float)DD));
  float ang = (float)n * div;
  acc += (d & 1) ? __cosf(ang) : __sinf(ang);
  h[idx] = acc;
}

__global__ __launch_bounds__(256) void pm_kernel(
    const float* __restrict__ mask, float* __restrict__ pm) {
  int m = blockIdx.x * 256 + threadIdx.x;
  if (m >= MM) return;
  int b = m >> 10, n = m & (NP - 1);
  int pr = n >> 5, pc = n & 31;
  float s = 0.f;
  for (int py = 0; py < PP; ++py)
#pragma unroll
    for (int px = 0; px < PP; ++px)
      s += mask[(((size_t)b * HW) + pr * PP + py) * HW + pc * PP + px];
  pm[m] = (s * (1.f / 64.f) > 0.5f) ? 1.f : 0.f;
}

// =====================================================================
// Y[M,N] = X[M,K] @ W[N,K]^T + bias.  W is bf16; X f32 or bf16; Y f32 or bf16.
// Tile 128 x BN x 32, double-buffered LDS; B (and bf16 A) staged via TDM.
// 256 threads = 8 waves as 4Mx2N; wave tile 32 x (BN/2).
// =====================================================================
#define TM 128
#define TK 32

template <typename TA, typename TO, int BN, int RELU>
__global__ __launch_bounds__(256) void gemm_kernel(
    const TA* __restrict__ X, const __bf16* __restrict__ W,
    const float* __restrict__ bias, TO* __restrict__ Y,
    int M, int N, int K) {
  constexpr int J = BN / 32;              // 16-col subtiles per wave
  __shared__ __align__(16) __bf16 As[2][TM * TK];
  __shared__ __align__(16) __bf16 Bs[2][BN * TK];

  const int tid  = threadIdx.x;
  const int wave = tid >> 5;
  const int wm   = wave >> 1;
  const int wn   = wave & 1;
  const int row0 = blockIdx.x * TM;
  const int col0 = blockIdx.y * BN;

  v8f acc[2][J] = {};

  auto stage = [&](int buf, int k0) {
#if USE_TDM
    if (wave == 0) {
      tdm_load_2d_bf16(lds_off(&Bs[buf][0]), W + (size_t)col0 * K + k0,
                       (unsigned)(K - k0), (unsigned)(N - col0), (unsigned)K,
                       TK, BN);
      if constexpr (std::is_same<TA, __bf16>::value) {
        tdm_load_2d_bf16(lds_off(&As[buf][0]), X + (size_t)row0 * K + k0,
                         (unsigned)(K - k0), (unsigned)(M - row0), (unsigned)K,
                         TK, TM);
      }
    }
#else
    for (int i = tid; i < BN * 4; i += 256) {        // 16B chunks of B
      int r = i >> 2, c8 = (i & 3) * 8;
      *reinterpret_cast<uint4*>(&Bs[buf][r * TK + c8]) =
          *reinterpret_cast<const uint4*>(W + (size_t)(col0 + r) * K + k0 + c8);
    }
    if constexpr (std::is_same<TA, __bf16>::value) {
      for (int i = tid; i < TM * 4; i += 256) {
        int r = i >> 2, c8 = (i & 3) * 8;
        *reinterpret_cast<uint4*>(&As[buf][r * TK + c8]) =
            *reinterpret_cast<const uint4*>(X + (size_t)(row0 + r) * K + k0 + c8);
      }
    }
#endif
    if constexpr (std::is_same<TA, float>::value) {
      // 128x32 f32 -> bf16 (1024 float4, 4 per thread)
#pragma unroll
      for (int i = 0; i < 4; ++i) {
        int f = tid + i * 256;
        int r = f >> 3, c = (f & 7) * 4;
        float4 v = *reinterpret_cast<const float4*>(
            X + (size_t)(row0 + r) * K + k0 + c);
        __bf16* d = &As[buf][r * TK + c];
        d[0] = (__bf16)v.x; d[1] = (__bf16)v.y;
        d[2] = (__bf16)v.z; d[3] = (__bf16)v.w;
      }
    }
  };

  auto compute = [&](int buf) {
    v16bf a0 = ld_frag(&As[buf][(wm * 32 +  0) * TK], TK);
    v16bf a1 = ld_frag(&As[buf][(wm * 32 + 16) * TK], TK);
#pragma unroll
    for (int j = 0; j < J; ++j) {
      v16bf b = ld_frag(&Bs[buf][(wn * (BN / 2) + j * 16) * TK], TK);
      acc[0][j] = wmma_bf16(a0, b, acc[0][j]);
      acc[1][j] = wmma_bf16(a1, b, acc[1][j]);
    }
  };

  stage(0, 0);
  if (wave == 0) tdm_wait();
  __syncthreads();

  const int nk = K / TK;
  for (int t = 0; t < nk; ++t) {
    int cur = t & 1;
    if (t + 1 < nk) stage(cur ^ 1, (t + 1) * TK);
    compute(cur);
    if (wave == 0) tdm_wait();
    __syncthreads();
  }

  const int lane = tid & 31;
  const int mofs = (lane >> 4) * 8;
  const int nn   = lane & 15;
#pragma unroll
  for (int i = 0; i < 2; ++i)
#pragma unroll
    for (int j = 0; j < J; ++j) {
      int col = col0 + wn * (BN / 2) + j * 16 + nn;
      float bv = bias[col];
#pragma unroll
      for (int r = 0; r < 8; ++r) {
        int rowg = row0 + wm * 32 + i * 16 + mofs + r;
        float v = acc[i][j][r] + bv;
        if (RELU) v = fmaxf(v, 0.f);
        Y[(size_t)rowg * N + col] = (TO)v;
      }
    }
}

// =====================================================================
// Flash attention on bf16 Q/K/V: grid (qtile=16, head=8, batch=4),
// 128 threads = 4 waves, wave owns 16 q-rows. Q/K tiles staged by TDM.
// =====================================================================
__global__ __launch_bounds__(128) void attn_kernel(
    const __bf16* __restrict__ Q, const __bf16* __restrict__ Kk,
    const __bf16* __restrict__ V, const float* __restrict__ pm,
    __bf16* __restrict__ O) {
  const int qt = blockIdx.x, hh = blockIdx.y, bb = blockIdx.z;
  __shared__ __align__(16) __bf16 Qs[64 * 64];
  __shared__ __align__(16) __bf16 Ks[64 * 64];
  __shared__ __align__(16) __bf16 Vt[64 * 64];     // transposed [d][k]
  __shared__ __align__(16) __bf16 Ps[4][16 * 64];

  const int tid = threadIdx.x;
  const int wave = tid >> 5;
  const int lane = tid & 31;
  const float scale = 0.125f;
  const size_t base = ((size_t)bb * NP) * DD + hh * DKK;

  // stage Q tile 64x64
#if USE_TDM
  if (wave == 0)
    tdm_load_2d_bf16(lds_off(Qs), Q + base + (size_t)qt * 64 * DD,
                     64, 64, DD, 64, 64);
  if (wave == 0) tdm_wait();
#else
#pragma unroll
  for (int i = 0; i < 4; ++i) {
    int f = tid + i * 128;
    int r = f >> 3, c8 = (f & 7) * 8;
    *reinterpret_cast<uint4*>(&Qs[r * 64 + c8]) =
        *reinterpret_cast<const uint4*>(Q + base + (size_t)(qt * 64 + r) * DD + c8);
  }
#endif
  __syncthreads();
  v16bf aQ0 = ld_frag(Qs + (wave * 16) * 64 +  0, 64);
  v16bf aQ1 = ld_frag(Qs + (wave * 16) * 64 + 32, 64);

  float rmax[8], rsum[8];
#pragma unroll
  for (int r = 0; r < 8; ++r) { rmax[r] = -1e30f; rsum[r] = 0.f; }
  v8f ctx[4] = {};

  const int mofs = (lane >> 4) * 8;
  const int nn   = lane & 15;

  for (int kt = 0; kt < 16; ++kt) {
    __syncthreads();
#if USE_TDM
    if (wave == 0)
      tdm_load_2d_bf16(lds_off(Ks), Kk + base + (size_t)kt * 64 * DD,
                       64, 64, DD, 64, 64);
#else
#pragma unroll
    for (int i = 0; i < 4; ++i) {
      int f = tid + i * 128;
      int r = f >> 3, c8 = (f & 7) * 8;
      *reinterpret_cast<uint4*>(&Ks[r * 64 + c8]) =
          *reinterpret_cast<const uint4*>(Kk + base + (size_t)(kt * 64 + r) * DD + c8);
    }
#endif
    // V tile transposed: each thread 4x(8 bf16) -> scattered b16 stores
#pragma unroll
    for (int i = 0; i < 4; ++i) {
      int f = tid + i * 128;
      int r = f >> 3, c8 = (f & 7) * 8;
      union { uint4 q; __bf16 h[8]; } u;
      u.q = *reinterpret_cast<const uint4*>(V + base + (size_t)(kt * 64 + r) * DD + c8);
#pragma unroll
      for (int e = 0; e < 8; ++e) Vt[(c8 + e) * 64 + r] = u.h[e];
    }
    if (wave == 0) tdm_wait();
    __syncthreads();

    // S = Q K^T
    v8f s[4];
#pragma unroll
    for (int j = 0; j < 4; ++j) {
      v8f z = {};
      v16bf bk0 = ld_frag(Ks + (j * 16) * 64 +  0, 64);
      v16bf bk1 = ld_frag(Ks + (j * 16) * 64 + 32, 64);
      z = wmma_bf16(aQ0, bk0, z);
      z = wmma_bf16(aQ1, bk1, z);
      s[j] = z;
    }

    // scale + key mask
#pragma unroll
    for (int j = 0; j < 4; ++j) {
      int col = kt * 64 + j * 16 + nn;
      float pmv = pm[bb * NP + col];
#pragma unroll
      for (int r = 0; r < 8; ++r)
        s[j][r] = (pmv > 0.5f) ? s[j][r] * scale : -1.0e9f;
    }

    float nm[8];
#pragma unroll
    for (int r = 0; r < 8; ++r) {
      float v = fmaxf(fmaxf(s[0][r], s[1][r]), fmaxf(s[2][r], s[3][r]));
      v = fmaxf(v, __shfl_xor(v, 1, 16));
      v = fmaxf(v, __shfl_xor(v, 2, 16));
      v = fmaxf(v, __shfl_xor(v, 4, 16));
      v = fmaxf(v, __shfl_xor(v, 8, 16));
      nm[r] = v;
    }
#pragma unroll
    for (int r = 0; r < 8; ++r) {
      float m2 = fmaxf(rmax[r], nm[r]);
      float corr = __expf(rmax[r] - m2);
      rmax[r] = m2;
      rsum[r] *= corr;
#pragma unroll
      for (int j = 0; j < 4; ++j) ctx[j][r] *= corr;
    }
#pragma unroll
    for (int j = 0; j < 4; ++j)
#pragma unroll
      for (int r = 0; r < 8; ++r)
        s[j][r] = __expf(s[j][r] - rmax[r]);
#pragma unroll
    for (int r = 0; r < 8; ++r) {
      float t = s[0][r] + s[1][r] + s[2][r] + s[3][r];
      t += __shfl_xor(t, 1, 16);
      t += __shfl_xor(t, 2, 16);
      t += __shfl_xor(t, 4, 16);
      t += __shfl_xor(t, 8, 16);
      rsum[r] += t;
    }

    // stash P (16x64) into this wave's LDS slot for PV
#pragma unroll
    for (int j = 0; j < 4; ++j)
#pragma unroll
      for (int r = 0; r < 8; ++r)
        Ps[wave][(mofs + r) * 64 + j * 16 + nn] = (__bf16)s[j][r];
    asm volatile("s_wait_dscnt 0" ::: "memory");

    v16bf aP0 = ld_frag(Ps[wave] +  0, 64);
    v16bf aP1 = ld_frag(Ps[wave] + 32, 64);
#pragma unroll
    for (int j = 0; j < 4; ++j) {
      v16bf bv0 = ld_frag(Vt + (j * 16) * 64 +  0, 64);
      v16bf bv1 = ld_frag(Vt + (j * 16) * 64 + 32, 64);
      ctx[j] = wmma_bf16(aP0, bv0, ctx[j]);
      ctx[j] = wmma_bf16(aP1, bv1, ctx[j]);
    }
  }

#pragma unroll
  for (int j = 0; j < 4; ++j)
#pragma unroll
    for (int r = 0; r < 8; ++r) {
      int rowg = qt * 64 + wave * 16 + mofs + r;
      O[base + (size_t)rowg * DD + j * 16 + nn] = (__bf16)(ctx[j][r] / rsum[r]);
    }
}

// =====================================================================
// Y = LayerNorm(X + R) * g + b ; one block (256 thr) per 512-wide row
// =====================================================================
__device__ __forceinline__ float blk_sum(float v, float* sm) {
#pragma unroll
  for (int o = 16; o > 0; o >>= 1) v += __shfl_xor(v, o, 32);
  int w = threadIdx.x >> 5;
  if ((threadIdx.x & 31) == 0) sm[w] = v;
  __syncthreads();
  float s = 0.f;
#pragma unroll
  for (int i = 0; i < 8; ++i) s += sm[i];
  __syncthreads();
  return s;
}

__global__ __launch_bounds__(256) void resln_kernel(
    const float* __restrict__ X, const float* __restrict__ R,
    const float* __restrict__ g, const float* __restrict__ b,
    float* __restrict__ Y) {
  __shared__ float sm[8];
  int row = blockIdx.x, tid = threadIdx.x;
  const float* xr = X + (size_t)row * DD;
  const float* rr = R + (size_t)row * DD;
  float v0 = xr[tid] + rr[tid];
  float v1 = xr[tid + 256] + rr[tid + 256];
  float mean = blk_sum(v0 + v1, sm) * (1.f / (float)DD);
  float d0 = v0 - mean, d1 = v1 - mean;
  float var = blk_sum(d0 * d0 + d1 * d1, sm) * (1.f / (float)DD);
  float rstd = rsqrtf(var + 1e-5f);
  Y[(size_t)row * DD + tid]       = d0 * rstd * g[tid]       + b[tid];
  Y[(size_t)row * DD + 256 + tid] = d1 * rstd * g[tid + 256] + b[tid + 256];
}

// =====================================================================
// un-patchify [4096,192] -> [4,3,256,256]
// =====================================================================
__global__ __launch_bounds__(256) void unpatch_kernel(
    const float* __restrict__ T, float* __restrict__ out) {
  size_t idx = (size_t)blockIdx.x * 256 + threadIdx.x;
  int ww = (int)(idx & 255);
  int hh = (int)((idx >> 8) & 255);
  int bc = (int)(idx >> 16);
  int b = bc / CCH, c = bc % CCH;
  int pr = hh >> 3, py = hh & 7, pc = ww >> 3, px = ww & 7;
  int n = pr * 32 + pc;
  out[idx] = T[((size_t)(b * NP + n)) * (CCH * PP * PP) + c * 64 + py * 8 + px];
}

// =====================================================================
extern "C" void kernel_launch(void* const* d_in, const int* in_sizes, int n_in,
                              void* d_out, int out_size, void* d_ws, size_t ws_size,
                              hipStream_t stream) {
  const float* x      = (const float*)d_in[0];
  const float* mask   = (const float*)d_in[1];
  const float* pw     = (const float*)d_in[2];
  const float* pb     = (const float*)d_in[3];
  const float* mw     = (const float*)d_in[4];
  const float* mbias  = (const float*)d_in[5];
  const float* Wq     = (const float*)d_in[6];
  const float* bq     = (const float*)d_in[7];
  const float* Wk     = (const float*)d_in[8];
  const float* bk     = (const float*)d_in[9];
  const float* Wv     = (const float*)d_in[10];
  const float* bv     = (const float*)d_in[11];
  const float* Wo     = (const float*)d_in[12];
  const float* bo     = (const float*)d_in[13];
  const float* g1     = (const float*)d_in[14];
  const float* be1    = (const float*)d_in[15];
  const float* W1     = (const float*)d_in[16];
  const float* b1     = (const float*)d_in[17];
  const float* W2     = (const float*)d_in[18];
  const float* b2     = (const float*)d_in[19];
  const float* g2     = (const float*)d_in[20];
  const float* be2    = (const float*)d_in[21];
  const float* out_w  = (const float*)d_in[22];
  const float* out_b  = (const float*)d_in[23];

  char* cur = (char*)d_ws;
  auto alloc = [&](size_t bytes) { char* p = cur; cur += (bytes + 255) & ~255ull; return p; };

  float*  h    = (float*) alloc((size_t)MM * DD * 4);
  float*  pmb  = (float*) alloc((size_t)MM * 4);
  __bf16* Qb   = (__bf16*)alloc((size_t)MM * DD * 2);
  __bf16* Kb   = (__bf16*)alloc((size_t)MM * DD * 2);
  __bf16* Vb   = (__bf16*)alloc((size_t)MM * DD * 2);
  __bf16* Ctx  = (__bf16*)alloc((size_t)MM * DD * 2);
  float*  T1   = (float*) alloc((size_t)MM * DD * 4);
  float*  Ab   = (float*) alloc((size_t)MM * DD * 4);
  __bf16* Fb   = (__bf16*)alloc((size_t)MM * DFF * 2);
  float*  T2   = (float*) alloc((size_t)MM * CCH * PP * PP * 4);
  __bf16* Wqb  = (__bf16*)alloc((size_t)LL * DD * DD * 2);
  __bf16* Wkb  = (__bf16*)alloc((size_t)LL * DD * DD * 2);
  __bf16* Wvb  = (__bf16*)alloc((size_t)LL * DD * DD * 2);
  __bf16* Wob  = (__bf16*)alloc((size_t)LL * DD * DD * 2);
  __bf16* W1b  = (__bf16*)alloc((size_t)LL * DFF * DD * 2);
  __bf16* W2b  = (__bf16*)alloc((size_t)LL * DD * DFF * 2);
  __bf16* Owb  = (__bf16*)alloc((size_t)CCH * PP * PP * DD * 2);

  // weight pre-conversion to bf16 (stays L2-resident: ~50MB on 192MB L2)
  const int nW  = LL * DD * DD / 4;    // 524288 float4
  const int nF  = LL * DFF * DD / 4;   // 2097152 float4
  const int nO  = CCH * PP * PP * DD / 4;
  cvt_bf16_kernel<<<(nW + 255) / 256, 256, 0, stream>>>(Wq, Wqb, nW);
  cvt_bf16_kernel<<<(nW + 255) / 256, 256, 0, stream>>>(Wk, Wkb, nW);
  cvt_bf16_kernel<<<(nW + 255) / 256, 256, 0, stream>>>(Wv, Wvb, nW);
  cvt_bf16_kernel<<<(nW + 255) / 256, 256, 0, stream>>>(Wo, Wob, nW);
  cvt_bf16_kernel<<<(nF + 255) / 256, 256, 0, stream>>>(W1, W1b, nF);
  cvt_bf16_kernel<<<(nF + 255) / 256, 256, 0, stream>>>(W2, W2b, nF);
  cvt_bf16_kernel<<<(nO + 255) / 256, 256, 0, stream>>>(out_w, Owb, nO);

  embed_kernel<<<(MM * DD) / 256, 256, 0, stream>>>(x, mask, pw, pb, mw, mbias, h);
  pm_kernel<<<MM / 256, 256, 0, stream>>>(mask, pmb);

  dim3 gProj(MM / TM, DD / 128);            // 32 x 4
  dim3 gFFN1(MM / TM, DFF / 128);           // 32 x 16
  dim3 gOut (MM / TM, (CCH * PP * PP) / 64);// 32 x 3
  dim3 gAttn(NP / 64, NHH, BQ);             // 16 x 8 x 4

  for (int l = 0; l < LL; ++l) {
    const __bf16* wq = Wqb + (size_t)l * DD * DD;
    const __bf16* wk = Wkb + (size_t)l * DD * DD;
    const __bf16* wv = Wvb + (size_t)l * DD * DD;
    const __bf16* wo = Wob + (size_t)l * DD * DD;
    const __bf16* w1 = W1b + (size_t)l * DFF * DD;
    const __bf16* w2 = W2b + (size_t)l * DD * DFF;

    gemm_kernel<float, __bf16, 128, 0><<<gProj, 256, 0, stream>>>(h, wq, bq + l * DD, Qb, MM, DD, DD);
    gemm_kernel<float, __bf16, 128, 0><<<gProj, 256, 0, stream>>>(h, wk, bk + l * DD, Kb, MM, DD, DD);
    gemm_kernel<float, __bf16, 128, 0><<<gProj, 256, 0, stream>>>(h, wv, bv + l * DD, Vb, MM, DD, DD);
    attn_kernel<<<gAttn, 128, 0, stream>>>(Qb, Kb, Vb, pmb, Ctx);
    gemm_kernel<__bf16, float, 128, 0><<<gProj, 256, 0, stream>>>(Ctx, wo, bo + l * DD, T1, MM, DD, DD);
    resln_kernel<<<MM, 256, 0, stream>>>(T1, h, g1 + l * DD, be1 + l * DD, Ab);
    gemm_kernel<float, __bf16, 128, 1><<<gFFN1, 256, 0, stream>>>(Ab, w1, b1 + l * DFF, Fb, MM, DFF, DD);
    gemm_kernel<__bf16, float, 128, 0><<<gProj, 256, 0, stream>>>(Fb, w2, b2 + l * DD, T1, MM, DD, DFF);
    resln_kernel<<<MM, 256, 0, stream>>>(T1, Ab, g2 + l * DD, be2 + l * DD, h);
  }

  gemm_kernel<float, float, 64, 0><<<gOut, 256, 0, stream>>>(h, Owb, out_b, T2, MM, CCH * PP * PP, DD);
  unpatch_kernel<<<(BQ * CCH * HW * HW) / 256, 256, 0, stream>>>(T2, (float*)d_out);
}
